// hourglass_mid_84920093377101
// MI455X (gfx1250) — matrix operand order repaired and checked by reading, not compiled, not run on hardware
//
#include <hip/hip_runtime.h>
#include <hip/hip_bf16.h>

typedef __attribute__((ext_vector_type(16))) _Float16 v16h;
typedef __attribute__((ext_vector_type(8)))  float    v8f;

// ---------------- GEMM modes ----------------
// All heavy ops are expressed as D[M,N] = gather(A)[M,K] * Wpacked[K,N] + bias (+residual)
//  SHIFT_H / SHIFT_W : token matmul with fused shift_chunk roll (zero at borders)
//  CONV_NCHW / CONV_TOK : implicit im2col 3x3 conv (pad 1), input NCHW or token layout
enum { GM_SHIFT_H = 0, GM_SHIFT_W = 1, GM_CONV_NCHW = 2, GM_CONV_TOK = 3 };

struct GemmParams {
    const float*    A;        // f32 activations
    const _Float16* Bp;       // packed f16 weights (WMMA B-fragment order)
    const float*    bias;     // [N]
    const float*    residual; // [M][N] tokens or nullptr
    float*          out;      // [M][N] tokens
    int M, N, K;
    int cin, in_h, in_w, stride; // geometry (conv) / token grid (shift)
};

// B packed layout: element j (0..15) of lane L of tile (kt,nt) at
//   Bp[(((size_t)kt*NT + nt)*32 + L)*16 + j]  ==  W^T[kt*32 + (L>>4)*16 + j][nt*16 + (L&15)]
template <int MODE, int NT>
__global__ __launch_bounds__(256) void wmma_gemm_kernel(GemmParams p) {
    const int lane  = threadIdx.x & 31;
    const int wave  = threadIdx.x >> 5;
    const int mtile = blockIdx.x * 8 + wave;
    const int m0    = mtile << 4;
    const int hf    = lane >> 4;      // wave half (A/B/C fragment layouts)
    const int l15   = lane & 15;
    const int mrow  = m0 + l15;       // A row owned by this lane
    const int kt_n  = p.K >> 5;

    // geometry of row mrow (token index -> (b, y, x))
    int Wo, HWo;
    if (MODE == GM_CONV_NCHW || MODE == GM_CONV_TOK) {
        const int Ho = (p.stride == 2) ? (p.in_h >> 1) : p.in_h;
        Wo  = (p.stride == 2) ? (p.in_w >> 1) : p.in_w;
        HWo = Ho * Wo;
    } else {
        Wo  = p.in_w;
        HWo = p.in_h * p.in_w;
    }
    const int bimg = mrow / HWo;
    const int rem  = mrow - bimg * HWo;
    const int ty   = rem / Wo;
    const int tx   = rem - ty * Wo;
    const int csz  = (p.K + 4) / 5;   // torch.chunk ceil-sized channel groups

    auto gather = [&](int kk) -> float {
        if (MODE == GM_SHIFT_H) {
            const int s  = kk / csz - 2;
            const int yy = ty - s;
            if ((unsigned)yy >= (unsigned)p.in_h) return 0.0f;
            return p.A[((size_t)bimg * HWo + (size_t)yy * Wo + tx) * p.K + kk];
        } else if (MODE == GM_SHIFT_W) {
            const int s  = kk / csz - 2;
            const int xx = tx - s;
            if ((unsigned)xx >= (unsigned)p.in_w) return 0.0f;
            return p.A[((size_t)bimg * HWo + (size_t)ty * Wo + xx) * p.K + kk];
        } else {
            const int cin = kk / 9;
            const int t   = kk - cin * 9;
            const int dy  = t / 3;
            const int dx  = t - dy * 3;
            const int iy  = ty * p.stride - 1 + dy;
            const int ix  = tx * p.stride - 1 + dx;
            if ((unsigned)iy >= (unsigned)p.in_h || (unsigned)ix >= (unsigned)p.in_w)
                return 0.0f;
            if (MODE == GM_CONV_NCHW)
                return p.A[(((size_t)bimg * p.cin + cin) * p.in_h + iy) * p.in_w + ix];
            else
                return p.A[((size_t)bimg * p.in_h * p.in_w + (size_t)iy * p.in_w + ix) * p.cin + cin];
        }
    };

    v8f acc[NT];
#pragma unroll
    for (int t = 0; t < NT; ++t) { v8f z = {}; acc[t] = z; }

    for (int kt = 0; kt < kt_n; ++kt) {
        const int k0 = kt << 5;
        v16h a;
#pragma unroll
        for (int i = 0; i < 8; ++i) {
            // A-layout (16x32 f16), matching packed B: k = half*16 + i*2
            const int kk = k0 + (hf << 4) + (i << 1);
            a[2 * i]     = (_Float16)gather(kk);
            a[2 * i + 1] = (_Float16)gather(kk + 1);
        }
        const _Float16* brow = p.Bp + (((size_t)kt * NT) * 32 + lane) * 16;
        __builtin_prefetch((const void*)(brow + (size_t)NT * 512), 0, 1); // next k-block
#pragma unroll
        for (int t = 0; t < NT; ++t) {
            v16h bf = *(const v16h*)(brow + (size_t)t * 512);
            acc[t] = __builtin_amdgcn_wmma_f32_16x16x32_f16(
                false, a, false, bf, (short)0, acc[t], false, false);
        }
    }

    // epilogue: bias + optional residual, C layout: row = m0 + half*8 + i, col = lane&15
#pragma unroll
    for (int t = 0; t < NT; ++t) {
        const int ncol = (t << 4) + l15;
        const float bv = p.bias[ncol];
#pragma unroll
        for (int i = 0; i < 8; ++i) {
            const int m = m0 + (hf << 3) + i;
            const size_t idx = (size_t)m * p.N + ncol;
            float v = acc[t][i] + bv;
            if (p.residual) v += p.residual[idx];
            p.out[idx] = v;
        }
    }
}

template <int MODE>
static inline void gemm_dispatch(const GemmParams& p, hipStream_t s) {
    dim3 grid(p.M / 128);
    switch (p.N >> 4) {
        case 8:  wmma_gemm_kernel<MODE, 8 ><<<grid, 256, 0, s>>>(p); break;
        case 10: wmma_gemm_kernel<MODE, 10><<<grid, 256, 0, s>>>(p); break;
        case 16: wmma_gemm_kernel<MODE, 16><<<grid, 256, 0, s>>>(p); break;
        default: break;
    }
}

// ---------------- weight packing (f32 -> f16, WMMA B-fragment order) ----------------
__global__ void pack_dense_w(const float* __restrict__ W, _Float16* __restrict__ out,
                             int N, int K) { // W: [N][K] (row-major), B = W^T
    const int idx = blockIdx.x * 256 + threadIdx.x;
    if (idx >= N * K) return;
    const int j    = idx & 15;
    const int lane = (idx >> 4) & 31;
    const int tile = idx >> 9;
    const int nt_n = N >> 4;
    const int nt   = tile % nt_n;
    const int kt   = tile / nt_n;
    const int n    = (nt << 4) + (lane & 15);
    const int k    = (kt << 5) + ((lane >> 4) << 4) + j;
    out[idx] = (_Float16)W[(size_t)n * K + k];
}

__global__ void pack_conv_w(const float* __restrict__ W, _Float16* __restrict__ out,
                            int Cout, int Cin) { // W: [Cout][Cin][3][3], K = Cin*9
    const int K   = Cin * 9;
    const int idx = blockIdx.x * 256 + threadIdx.x;
    if (idx >= Cout * K) return;
    const int j    = idx & 15;
    const int lane = (idx >> 4) & 31;
    const int tile = idx >> 9;
    const int nt_n = Cout >> 4;
    const int nt   = tile % nt_n;
    const int kt   = tile / nt_n;
    const int n    = (nt << 4) + (lane & 15);
    const int k    = (kt << 5) + ((lane >> 4) << 4) + j;
    const int cin  = k / 9;
    const int t    = k - cin * 9;
    out[idx] = (_Float16)W[(((size_t)n * Cin + cin) * 3 + t / 3) * 3 + (t % 3)];
}

// ---------------- layernorm over channels: rows = B*N tokens, one wave32 per row ----
__global__ __launch_bounds__(256) void layernorm_kernel(
    const float* __restrict__ x, const float* __restrict__ g, const float* __restrict__ b,
    float* __restrict__ out, int rows, int C) {
    const int row  = blockIdx.x * 8 + (threadIdx.x >> 5);
    const int lane = threadIdx.x & 31;
    if (row >= rows) return;
    const float* xr = x + (size_t)row * C;
    float s = 0.f, s2 = 0.f;
    for (int c = lane; c < C; c += 32) { const float v = xr[c]; s += v; s2 += v * v; }
    for (int off = 16; off > 0; off >>= 1) {
        s  += __shfl_down(s, off, 32);
        s2 += __shfl_down(s2, off, 32);
    }
    s  = __shfl(s, 0, 32);
    s2 = __shfl(s2, 0, 32);
    const float mu  = s / C;
    const float inv = rsqrtf(s2 / C - mu * mu + 1e-5f);
    float* yr = out + (size_t)row * C;
    for (int c = lane; c < C; c += 32) yr[c] = (xr[c] - mu) * inv * g[c] + b[c];
}

// ---------------- depthwise 3x3 conv + exact GELU, token layout -------------------
__global__ void dwconv_gelu_kernel(const float* __restrict__ h, const float* __restrict__ w,
                                   const float* __restrict__ bias, float* __restrict__ out,
                                   int B, int C, int H, int W) {
    const size_t total = (size_t)B * H * W * C;
    const size_t idx   = (size_t)blockIdx.x * 256 + threadIdx.x;
    if (idx >= total) return;
    const int c = (int)(idx % C);
    size_t r    = idx / C;
    const int x = (int)(r % W); r /= W;
    const int y = (int)(r % H);
    const int b = (int)(r / H);
    const float* hb = h + (size_t)b * H * W * C;
    float acc = bias[c];
#pragma unroll
    for (int dy = 0; dy < 3; ++dy) {
#pragma unroll
        for (int dx = 0; dx < 3; ++dx) {
            const int iy = y + dy - 1, ix = x + dx - 1;
            if ((unsigned)iy < (unsigned)H && (unsigned)ix < (unsigned)W)
                acc += hb[((size_t)iy * W + ix) * C + c] * w[c * 9 + dy * 3 + dx];
        }
    }
    out[idx] = 0.5f * acc * (1.0f + erff(acc * 0.70710678118654752f));
}

// ---------------- BN (running stats) + bilinear 2x upsample + ReLU + skip add -----
// BN is per-channel affine => commutes with bilinear interp; conv input is tokens.
__global__ void bn_up_relu_add_kernel(const float* __restrict__ conv,
                                      const float* __restrict__ g, const float* __restrict__ bb,
                                      const float* __restrict__ rm, const float* __restrict__ rv,
                                      const float* __restrict__ skip, int skip_is_tokens,
                                      float* __restrict__ out, int B, int C, int Hs, int Ws) {
    const int H = Hs * 2, W = Ws * 2;
    const size_t total = (size_t)B * H * W * C;
    const size_t idx   = (size_t)blockIdx.x * 256 + threadIdx.x;
    if (idx >= total) return;
    const int c = (int)(idx % C);
    size_t r    = idx / C;
    const int x = (int)(r % W); r /= W;
    const int y = (int)(r % H);
    const int b = (int)(r / H);
    const float fy = (y + 0.5f) * 0.5f - 0.5f;
    const float fx = (x + 0.5f) * 0.5f - 0.5f;
    const int y0i = (int)floorf(fy), x0i = (int)floorf(fx);
    const float wy = fy - y0i, wx = fx - x0i;
    const int y0 = min(max(y0i, 0), Hs - 1), y1 = min(max(y0i + 1, 0), Hs - 1);
    const int x0 = min(max(x0i, 0), Ws - 1), x1 = min(max(x0i + 1, 0), Ws - 1);
    const float* cb = conv + (size_t)b * Hs * Ws * C;
    const float v00 = cb[((size_t)y0 * Ws + x0) * C + c];
    const float v01 = cb[((size_t)y0 * Ws + x1) * C + c];
    const float v10 = cb[((size_t)y1 * Ws + x0) * C + c];
    const float v11 = cb[((size_t)y1 * Ws + x1) * C + c];
    float val = v00 * (1 - wy) * (1 - wx) + v01 * (1 - wy) * wx +
                v10 * wy * (1 - wx)       + v11 * wy * wx;
    val = (val - rm[c]) * (g[c] * rsqrtf(rv[c] + 1e-5f)) + bb[c];
    val = fmaxf(val, 0.0f);
    const float sk = skip_is_tokens ? skip[idx]
                                    : skip[(((size_t)b * C + c) * H + y) * W + x];
    out[idx] = val + sk;
}

// ---------------- tokens (B,HW,C) -> NCHW ----------------
__global__ void tok2nchw_kernel(const float* __restrict__ tok, float* __restrict__ out,
                                int B, int C, int H, int W) {
    const size_t total = (size_t)B * C * H * W;
    const size_t idx   = (size_t)blockIdx.x * 256 + threadIdx.x;
    if (idx >= total) return;
    const int x = (int)(idx % W);
    size_t r    = idx / W;
    const int y = (int)(r % H); r /= H;
    const int c = (int)(r % C);
    const int b = (int)(r / C);
    out[idx] = tok[((size_t)b * H * W + (size_t)y * W + x) * C + c];
}

// =================================================================================
extern "C" void kernel_launch(void* const* d_in, const int* in_sizes, int n_in,
                              void* d_out, int out_size, void* d_ws, size_t ws_size,
                              hipStream_t stream) {
    (void)in_sizes; (void)n_in; (void)out_size; (void)ws_size;
    const float* x = (const float*)d_in[0];
    auto F = [&](int i) { return (const float*)d_in[i]; };

    // ---- workspace layout (floats) ----
    const size_t BIG = 33554432ull;              // 4*256*256*128 tokens
    float* U0 = (float*)d_ws;
    float* Ua = U0 + BIG;
    float* Ub = Ua + BIG;
    float* U1 = Ub + BIG;                        // t4 tokens: 65536*160
    _Float16* pw = (_Float16*)(U1 + 10485760ull);
    _Float16* pw_pe3  = pw;                 // 1152*160
    _Float16* pw_pe4  = pw_pe3  + 184320;   // 1440*256
    _Float16* pw_d1   = pw_pe4  + 368640;   // 2304*160
    _Float16* pw_d2   = pw_d1   + 368640;   // 1440*128
    _Float16* pw_b1w1 = pw_d2   + 184320;
    _Float16* pw_b1w2 = pw_b1w1 + 25600;
    _Float16* pw_b2w1 = pw_b1w2 + 25600;
    _Float16* pw_b2w2 = pw_b2w1 + 65536;
    _Float16* pw_db1w1 = pw_b2w2 + 65536;
    _Float16* pw_db1w2 = pw_db1w1 + 25600;
    _Float16* pw_db2w1 = pw_db1w2 + 25600;
    _Float16* pw_db2w2 = pw_db2w1 + 16384;

    // ---- pack all weights to f16 WMMA fragment order ----
    auto packc = [&](const float* w, _Float16* o, int co, int ci) {
        pack_conv_w<<<(co * ci * 9 + 255) / 256, 256, 0, stream>>>(w, o, co, ci);
    };
    auto packd = [&](const float* w, _Float16* o, int n, int k) {
        pack_dense_w<<<(n * k + 255) / 256, 256, 0, stream>>>(w, o, n, k);
    };
    packc(F(1),  pw_pe3, 160, 128);
    packc(F(5),  pw_pe4, 256, 160);
    packc(F(49), pw_d1,  160, 256);
    packc(F(51), pw_d2,  128, 160);
    packd(F(11), pw_b1w1, 160, 160);  packd(F(15), pw_b1w2, 160, 160);
    packd(F(19), pw_b2w1, 256, 256);  packd(F(23), pw_b2w2, 256, 256);
    packd(F(27), pw_db1w1, 160, 160); packd(F(31), pw_db1w2, 160, 160);
    packd(F(35), pw_db2w1, 128, 128); packd(F(39), pw_db2w2, 128, 128);

    // shifted block: out(tmpB) = xin + W2 @ shiftW(gelu(dw(W1 @ shiftH(LN(xin)))))
    auto run_block = [&](float* xin, float* tA, float* tB, int M, int C, int H, int W,
                         int ing, int inb, const _Float16* w1, int ib1,
                         int idw, int idb, const _Float16* w2, int ib2) {
        layernorm_kernel<<<M / 8, 256, 0, stream>>>(xin, F(ing), F(inb), tA, M, C);
        GemmParams p{};
        p.A = tA; p.Bp = w1; p.bias = F(ib1); p.residual = nullptr; p.out = tB;
        p.M = M; p.N = C; p.K = C; p.in_h = H; p.in_w = W; p.stride = 1; p.cin = C;
        gemm_dispatch<GM_SHIFT_H>(p, stream);
        const size_t tot = (size_t)M * C;
        dwconv_gelu_kernel<<<(unsigned)((tot + 255) / 256), 256, 0, stream>>>(
            tB, F(idw), F(idb), tA, M / (H * W), C, H, W);
        p.A = tA; p.Bp = w2; p.bias = F(ib2); p.residual = xin; p.out = tB;
        gemm_dispatch<GM_SHIFT_W>(p, stream);
    };

    GemmParams p{};

    // ======== encoder stage 3: patch_embed pe3 (128->160, /2) + block b1 ========
    p.A = x; p.Bp = pw_pe3; p.bias = F(2); p.residual = nullptr; p.out = U0;
    p.M = 65536; p.N = 160; p.K = 1152; p.cin = 128; p.in_h = 256; p.in_w = 256; p.stride = 2;
    gemm_dispatch<GM_CONV_NCHW>(p, stream);
    layernorm_kernel<<<8192, 256, 0, stream>>>(U0, F(3), F(4), U1, 65536, 160);
    run_block(U1, U0, Ua, 65536, 160, 128, 128, 9, 10, pw_b1w1, 12, 13, 14, pw_b1w2, 16);
    layernorm_kernel<<<8192, 256, 0, stream>>>(Ua, F(41), F(42), U1, 65536, 160); // U1 = t4

    // ======== encoder stage 4: patch_embed pe4 (160->256, /2) + block b2 ========
    p = GemmParams{};
    p.A = U1; p.Bp = pw_pe4; p.bias = F(6); p.out = U0;
    p.M = 16384; p.N = 256; p.K = 1440; p.cin = 160; p.in_h = 128; p.in_w = 128; p.stride = 2;
    gemm_dispatch<GM_CONV_TOK>(p, stream);
    layernorm_kernel<<<2048, 256, 0, stream>>>(U0, F(7), F(8), Ua, 16384, 256);
    run_block(Ua, U0, Ub, 16384, 256, 64, 64, 17, 18, pw_b2w1, 20, 21, 22, pw_b2w2, 24);
    layernorm_kernel<<<2048, 256, 0, stream>>>(Ub, F(43), F(44), U0, 16384, 256);

    // ======== decoder 1: conv d1 (256->160) + BN + up2x + ReLU + t4 + block db1 ========
    p = GemmParams{};
    p.A = U0; p.Bp = pw_d1; p.bias = F(50); p.out = Ua;
    p.M = 16384; p.N = 160; p.K = 2304; p.cin = 256; p.in_h = 64; p.in_w = 64; p.stride = 1;
    gemm_dispatch<GM_CONV_TOK>(p, stream);
    bn_up_relu_add_kernel<<<(unsigned)(((size_t)65536 * 160 + 255) / 256), 256, 0, stream>>>(
        Ua, F(53), F(54), F(55), F(56), U1, 1, Ub, 4, 160, 64, 64);
    run_block(Ub, U0, Ua, 65536, 160, 128, 128, 25, 26, pw_db1w1, 28, 29, 30, pw_db1w2, 32);
    layernorm_kernel<<<8192, 256, 0, stream>>>(Ua, F(45), F(46), U0, 65536, 160);

    // ======== decoder 2: conv d2 (160->128) + BN + up2x + ReLU + x + block db2 ========
    p = GemmParams{};
    p.A = U0; p.Bp = pw_d2; p.bias = F(52); p.out = Ua;
    p.M = 65536; p.N = 128; p.K = 1440; p.cin = 160; p.in_h = 128; p.in_w = 128; p.stride = 1;
    gemm_dispatch<GM_CONV_TOK>(p, stream);
    bn_up_relu_add_kernel<<<(unsigned)(((size_t)262144 * 128 + 255) / 256), 256, 0, stream>>>(
        Ua, F(57), F(58), F(59), F(60), x, 0, Ub, 4, 128, 128, 128);
    run_block(Ub, U0, Ua, 262144, 128, 256, 256, 33, 34, pw_db2w1, 36, 37, 38, pw_db2w2, 40);
    layernorm_kernel<<<32768, 256, 0, stream>>>(Ua, F(47), F(48), U0, 262144, 128);

    // ======== final: tokens -> NCHW output ========
    tok2nchw_kernel<<<(unsigned)(((size_t)262144 * 128 + 255) / 256), 256, 0, stream>>>(
        U0, (float*)d_out, 4, 128, 256, 256);
}
